// SetAbstractionLayer_34763465294633
// MI455X (gfx1250) — matrix operand order repaired and checked by reading, not compiled, not run on hardware
//
#include <hip/hip_runtime.h>
#include <hip/hip_bf16.h>

// ---- problem constants (must match reference) ----
#define B_C   4
#define NPB_C 8192
#define M_C   2048
#define K_C   64          // max neighbors
#define F_C   64          // feature dim
#define R2_C  0.04f       // RADIUS^2

typedef __attribute__((ext_vector_type(16))) _Float16 v16h;
typedef __attribute__((ext_vector_type(8)))  _Float16 v8h;
typedef __attribute__((ext_vector_type(4)))  _Float16 v4h;
typedef __attribute__((ext_vector_type(8)))  float    v8f;

// =====================================================================
// 1) Farthest point sampling: one block per cloud, 1024 threads,
//    8 points per thread held in registers; LDS tree argmax per round.
// =====================================================================
#define FPT 8  // NPB_C / 1024

__global__ __launch_bounds__(1024) void fps_kernel(const float* __restrict__ pos,
                                                   int* __restrict__ samp) {
  __shared__ float rv[1024];
  __shared__ int   ri[1024];
  __shared__ float bpt[3];

  const int b   = blockIdx.x;
  const int tid = threadIdx.x;
  const float* pb = pos + (size_t)b * NPB_C * 3;

  float px[FPT], py[FPT], pz[FPT], md[FPT];
  const float x0 = pb[0], y0 = pb[1], z0 = pb[2];
#pragma unroll
  for (int i = 0; i < FPT; ++i) {
    const int p = i * 1024 + tid;
    px[i] = pb[(size_t)p * 3 + 0];
    py[i] = pb[(size_t)p * 3 + 1];
    pz[i] = pb[(size_t)p * 3 + 2];
    const float dx = px[i] - x0, dy = py[i] - y0, dz = pz[i] - z0;
    md[i] = dx * dx + dy * dy + dz * dz;
  }
  if (tid == 0) samp[(size_t)b * M_C] = 0;

  for (int it = 1; it < M_C; ++it) {
    // local argmax (first-index tie-break, matching jnp.argmax)
    float bv = -__builtin_inff();
    int   bi = 0x7fffffff;
#pragma unroll
    for (int i = 0; i < FPT; ++i) {
      const int p = i * 1024 + tid;
      if (md[i] > bv) { bv = md[i]; bi = p; }
    }
    rv[tid] = bv; ri[tid] = bi;
    __syncthreads();
    for (int st = 512; st > 0; st >>= 1) {
      if (tid < st) {
        const float ov = rv[tid + st]; const int oi = ri[tid + st];
        if (ov > rv[tid] || (ov == rv[tid] && oi < ri[tid])) { rv[tid] = ov; ri[tid] = oi; }
      }
      __syncthreads();
    }
    if (tid == 0) {
      const int best = ri[0];
      samp[(size_t)b * M_C + it] = best;
      bpt[0] = pb[(size_t)best * 3 + 0];
      bpt[1] = pb[(size_t)best * 3 + 1];
      bpt[2] = pb[(size_t)best * 3 + 2];
    }
    __syncthreads();
    const float bx = bpt[0], by = bpt[1], bz = bpt[2];
#pragma unroll
    for (int i = 0; i < FPT; ++i) {
      const float dx = px[i] - bx, dy = py[i] - by, dz = pz[i] - bz;
      const float d = dx * dx + dy * dy + dz * dz;
      md[i] = fminf(md[i], d);
    }
    __syncthreads();
  }
}

// =====================================================================
// 2) Radius ball query + exact top-K (ascending (d2, idx)).
//    One block per sample; d2 row cached in LDS (32 KB).
//    Also writes pos_out and batch_out.
// =====================================================================
__global__ __launch_bounds__(256) void ball_kernel(const float* __restrict__ pos,
                                                   const int* __restrict__ samp,
                                                   int* __restrict__ nbr,
                                                   float* __restrict__ pos_out,
                                                   int* __restrict__ batch_out) {
  __shared__ float d2s[NPB_C];    // 32 KB
  __shared__ float rv[256];
  __shared__ int   ri[256];
  __shared__ float last_d;
  __shared__ int   last_i;

  const int s   = blockIdx.x;          // sample 0..B*M-1
  const int b   = s / M_C;
  const int tid = threadIdx.x;

  const int ci = samp[s];
  const float cx = pos[((size_t)b * NPB_C + ci) * 3 + 0];
  const float cy = pos[((size_t)b * NPB_C + ci) * 3 + 1];
  const float cz = pos[((size_t)b * NPB_C + ci) * 3 + 2];

  for (int p = tid; p < NPB_C; p += 256) {
    const float* pp = pos + ((size_t)b * NPB_C + p) * 3;
    const float dx = pp[0] - cx, dy = pp[1] - cy, dz = pp[2] - cz;
    d2s[p] = dx * dx + dy * dy + dz * dz;
  }
  if (tid == 0) { last_d = -1.0f; last_i = -1; }
  __syncthreads();

  for (int k = 0; k < K_C; ++k) {
    const float ld = last_d; const int li = last_i;
    float bv = __builtin_inff(); int bi = 0x7fffffff;
    for (int p = tid; p < NPB_C; p += 256) {
      const float d = d2s[p];
      const bool after = (d > ld) || (d == ld && p > li);
      if (after && (d < bv || (d == bv && p < bi))) { bv = d; bi = p; }
    }
    rv[tid] = bv; ri[tid] = bi;
    __syncthreads();
    for (int st = 128; st > 0; st >>= 1) {
      if (tid < st) {
        const float ov = rv[tid + st]; const int oi = ri[tid + st];
        if (ov < rv[tid] || (ov == rv[tid] && oi < ri[tid])) { rv[tid] = ov; ri[tid] = oi; }
      }
      __syncthreads();
    }
    if (tid == 0) {
      const float bd = rv[0]; const int bb = ri[0];
      const bool ok = (bb != 0x7fffffff) && (bd <= R2_C);
      nbr[(size_t)s * K_C + k] = ok ? bb : -1;
      last_d = bd; last_i = bb;
    }
    __syncthreads();
  }
  if (tid == 0) {
    pos_out[(size_t)s * 3 + 0] = cx;
    pos_out[(size_t)s * 3 + 1] = cy;
    pos_out[(size_t)s * 3 + 2] = cz;
    batch_out[s] = b;
  }
}

// =====================================================================
// 3) Weight prep: fp32 -> fp16, pre-swizzled into WMMA B-fragment
//    order: one contiguous 16-half (32 B) record per (fragment, lane).
//    fragment id f = nt*KC32 + kc; element i of lane L maps to
//    W[kc*32 + (i<8 ? (L>>4)*8 + i : 16 + (L>>4)*8 + i-8)][nt*16 + (L&15)].
// =====================================================================
__global__ void prep_kernel(const float* __restrict__ W1, const float* __restrict__ W2,
                            const float* __restrict__ W3,
                            _Float16* __restrict__ Wf1, _Float16* __restrict__ Wf2,
                            _Float16* __restrict__ Wf3) {
  const int tid = threadIdx.x;
  const int stride = blockDim.x;
  // layer 1: KDIM=96 (rows 67..95 zero), NCOL=64 -> 4 ntiles x 3 kc
  for (int e = tid; e < 4 * 3 * 32 * 16; e += stride) {
    const int i = e & 15, lane = (e >> 4) & 31, f = e >> 9;
    const int nt = f / 3, kc = f % 3;
    const int k = kc * 32 + (i < 8 ? (lane >> 4) * 8 + i : 16 + (lane >> 4) * 8 + (i - 8));
    const int n = nt * 16 + (lane & 15);
    Wf1[e] = (_Float16)(k < 67 ? W1[k * 64 + n] : 0.0f);
  }
  // layer 2: KDIM=64, NCOL=64 -> 4 ntiles x 2 kc
  for (int e = tid; e < 4 * 2 * 32 * 16; e += stride) {
    const int i = e & 15, lane = (e >> 4) & 31, f = e >> 9;
    const int nt = f >> 1, kc = f & 1;
    const int k = kc * 32 + (i < 8 ? (lane >> 4) * 8 + i : 16 + (lane >> 4) * 8 + (i - 8));
    const int n = nt * 16 + (lane & 15);
    Wf2[e] = (_Float16)W2[k * 64 + n];
  }
  // layer 3: KDIM=64, NCOL=128 -> 8 ntiles x 2 kc
  for (int e = tid; e < 8 * 2 * 32 * 16; e += stride) {
    const int i = e & 15, lane = (e >> 4) & 31, f = e >> 9;
    const int nt = f >> 1, kc = f & 1;
    const int k = kc * 32 + (i < 8 ? (lane >> 4) * 8 + i : 16 + (lane >> 4) * 8 + (i - 8));
    const int n = nt * 16 + (lane & 15);
    Wf3[e] = (_Float16)W3[k * 128 + n];
  }
}

// =====================================================================
// 4) WMMA MLP + masked max-pool.  One block (128 thr = 4 waves) per
//    sample; wave w owns neighbor-row tile w (rows 16w..16w+15).
// =====================================================================

// A fragment: 16x32 f16 from LDS row-major A (stride in halves).
// Elements 0..7 and 8..15 are contiguous runs -> two ds_load_b128.
__device__ inline v16h load_a_frag(const _Float16* __restrict__ A, int strideA,
                                   int mt, int kc, int lane) {
  const int m  = mt * 16 + (lane & 15);
  const int kb = kc * 32 + (lane >> 4) * 8;
  const v8h lo = *(const v8h*)(A + m * strideA + kb);
  const v8h hi = *(const v8h*)(A + m * strideA + kb + 16);
  return __builtin_shufflevector(lo, hi, 0, 1, 2, 3, 4, 5, 6, 7,
                                 8, 9, 10, 11, 12, 13, 14, 15);
}

// B fragment: one contiguous 32-byte record per (fragment, lane) from
// the pre-swizzled global weight array (L2/WGP$-resident).
__device__ inline v16h load_b_frag(const _Float16* __restrict__ Wf, int fragid, int lane) {
  return *(const v16h*)(Wf + ((size_t)(fragid * 32 + lane) << 4));
}

// Hidden layer: Hout = relu(A @ W + bias) in f16.  A fragments hoisted
// out of the ntile loop; B fragments streamed from global.
template <int KC32, int NT>
__device__ inline void gemm_layer(const _Float16* __restrict__ A, int strideA,
                                  const _Float16* __restrict__ Wf,
                                  const float* __restrict__ bias_g,
                                  _Float16* __restrict__ Hout,
                                  int lane, int wid) {
  const int mt = wid;                       // 4 waves -> 4 row tiles
  v16h afr[KC32];
#pragma unroll
  for (int kc = 0; kc < KC32; ++kc) afr[kc] = load_a_frag(A, strideA, mt, kc, lane);

  const int mrow = mt * 16 + (lane >> 4) * 8;
  for (int nt = 0; nt < NT; ++nt) {
    v8f acc = {};
#pragma unroll
    for (int kc = 0; kc < KC32; ++kc) {
      const v16h bf = load_b_frag(Wf, nt * KC32 + kc, lane);
      acc = __builtin_amdgcn_wmma_f32_16x16x32_f16(false, afr[kc], false, bf,
                                                   (short)0, acc, false, false);
    }
    const int n = nt * 16 + (lane & 15);
    const float bias = bias_g[n];
#pragma unroll
    for (int v = 0; v < 8; ++v) {
      float h = acc[v] + bias;
      Hout[(mrow + v) * (NT * 16) + n] = (_Float16)(h > 0.0f ? h : 0.0f);
    }
  }
}

__global__ __launch_bounds__(128) void mlp_kernel(const float* __restrict__ x,
                                                  const float* __restrict__ pos,
                                                  const int* __restrict__ nbr,
                                                  const int* __restrict__ samp,
                                                  const _Float16* __restrict__ Wf1,
                                                  const _Float16* __restrict__ Wf2,
                                                  const _Float16* __restrict__ Wf3,
                                                  const float* __restrict__ b1,
                                                  const float* __restrict__ b2,
                                                  const float* __restrict__ b3,
                                                  float* __restrict__ out) {
  __shared__ __align__(16) _Float16 Af[64 * 96];  // layer-1 A (67 -> 96)  12 KB
  __shared__ __align__(16) _Float16 H1[64 * 64];  //  8 KB
  __shared__ __align__(16) _Float16 H2[64 * 64];  //  8 KB
  __shared__ float    outmax[128];
  __shared__ int      rowvalid[64];
  __shared__ float    ctr[3];

  const int s    = blockIdx.x;         // sample id
  const int b    = s / M_C;
  const int tid  = threadIdx.x;
  const int lane = tid & 31;
  const int wid  = tid >> 5;

  // center position + pooling init + row validity
  const int ci = samp[s];
  if (tid < 3)  ctr[tid] = pos[((size_t)b * NPB_C + ci) * 3 + tid];
  outmax[tid] = -__builtin_inff();     // 128 threads == 128 outputs
  if (tid < 64) rowvalid[tid] = (nbr[(size_t)s * K_C + tid] >= 0) ? 1 : 0;
  __syncthreads();

  // gather: Af[r][0:64]=x[j] (float4 loads), [64:67]=pos[j]-center, [67:96]=0
  for (int e = tid; e < 64 * 24; e += 128) {       // 24 chunks of 4 halves
    const int r = e / 24, c4 = e % 24;
    int j = nbr[(size_t)s * K_C + r];
    if (j < 0) j = 0;                  // reference also gathers for masked rows
    const size_t gp = (size_t)b * NPB_C + j;
    v4h o;
    if (c4 < 16) {
      const float4 v = ((const float4*)(x + gp * 64))[c4];
      o[0] = (_Float16)v.x; o[1] = (_Float16)v.y;
      o[2] = (_Float16)v.z; o[3] = (_Float16)v.w;
    } else {
#pragma unroll
      for (int q = 0; q < 4; ++q) {
        const int c = c4 * 4 + q;
        float v = 0.0f;
        if (c < 67) v = pos[gp * 3 + (c - 64)] - ctr[c - 64];
        o[q] = (_Float16)v;
      }
    }
    *(v4h*)(Af + r * 96 + c4 * 4) = o;
  }
  __syncthreads();

  // ---- layer 1: [64x96] @ [96x64] ----
  gemm_layer<3, 4>(Af, 96, Wf1, b1, H1, lane, wid);
  __syncthreads();

  // ---- layer 2: [64x64] @ [64x64] ----
  gemm_layer<2, 4>(H1, 64, Wf2, b2, H2, lane, wid);
  __syncthreads();

  // ---- layer 3: [64x64] @ [64x128] + masked max-pool over rows ----
  {
    const int mt = wid;
    v16h afr[2];
#pragma unroll
    for (int kc = 0; kc < 2; ++kc) afr[kc] = load_a_frag(H2, 64, mt, kc, lane);
    const int mrow = mt * 16 + (lane >> 4) * 8;
    for (int nt = 0; nt < 8; ++nt) {
      v8f acc = {};
#pragma unroll
      for (int kc = 0; kc < 2; ++kc) {
        const v16h bf = load_b_frag(Wf3, nt * 2 + kc, lane);
        acc = __builtin_amdgcn_wmma_f32_16x16x32_f16(false, afr[kc], false, bf,
                                                     (short)0, acc, false, false);
      }
      const int n = nt * 16 + (lane & 15);
      const float bias = b3[n];
      float mx = -__builtin_inff();
      bool any = false;
#pragma unroll
      for (int v = 0; v < 8; ++v) {
        if (rowvalid[mrow + v]) {
          float h = acc[v] + bias;
          h = h > 0.0f ? h : 0.0f;     // relu -> value >= 0
          mx = fmaxf(mx, h);
          any = true;
        }
      }
      // post-relu values are >= 0 so signed-int compare == float compare;
      // -inf init has sign bit set and stays smallest.
      if (any) atomicMax((int*)&outmax[n], __float_as_int(mx));
    }
  }
  __syncthreads();
  out[(size_t)s * 128 + tid] = outmax[tid];
}

// =====================================================================
// launcher
// =====================================================================
static inline size_t align256(size_t v) { return (v + 255) & ~(size_t)255; }

extern "C" void kernel_launch(void* const* d_in, const int* in_sizes, int n_in,
                              void* d_out, int out_size, void* d_ws, size_t ws_size,
                              hipStream_t stream) {
  const float* x   = (const float*)d_in[0];
  const float* pos = (const float*)d_in[1];
  // d_in[2] = batch (unused; regenerated)
  const float* W1 = (const float*)d_in[3];
  const float* b1 = (const float*)d_in[4];
  const float* W2 = (const float*)d_in[5];
  const float* b2 = (const float*)d_in[6];
  const float* W3 = (const float*)d_in[7];
  const float* b3 = (const float*)d_in[8];

  const size_t BM = (size_t)B_C * M_C;

  char* ws = (char*)d_ws;
  int* samp = (int*)ws;           ws += align256(BM * sizeof(int));
  int* nbr  = (int*)ws;           ws += align256(BM * K_C * sizeof(int));
  _Float16* Wf1 = (_Float16*)ws;  ws += align256(4 * 3 * 32 * 16 * sizeof(_Float16));
  _Float16* Wf2 = (_Float16*)ws;  ws += align256(4 * 2 * 32 * 16 * sizeof(_Float16));
  _Float16* Wf3 = (_Float16*)ws;  ws += align256(8 * 2 * 32 * 16 * sizeof(_Float16));

  float* out       = (float*)d_out;                // [B*M, 128]
  float* pos_out   = out + BM * 128;               // [B*M, 3]
  int*   batch_out = (int*)(pos_out + BM * 3);     // [B*M] (int32 bit pattern)

  fps_kernel<<<B_C, 1024, 0, stream>>>(pos, samp);
  prep_kernel<<<1, 256, 0, stream>>>(W1, W2, W3, Wf1, Wf2, Wf3);
  ball_kernel<<<(int)BM, 256, 0, stream>>>(pos, samp, nbr, pos_out, batch_out);
  mlp_kernel<<<(int)BM, 128, 0, stream>>>(x, pos, nbr, samp, Wf1, Wf2, Wf3,
                                          b1, b2, b3, out);
}
